// GCN_19095424598296
// MI455X (gfx1250) — matrix operand order repaired and checked by reading, not compiled, not run on hardware
//
#include <hip/hip_runtime.h>
#include <hip/hip_bf16.h>

// ---------------------------------------------------------------------------
// GCN on MI455X (gfx1250, wave32):
//   - dense transforms via v_wmma_f32_16x16x32_bf16, W staged in LDS (bf16,
//     fragment order) -> B frag = one 32B ds_load; A frag reused across 8 tiles
//   - relu is a compile-time template flag (no per-element cndmask blending)
//   - edge aggregation via coalesced float4 gathers + global_atomic_add_f32
// ---------------------------------------------------------------------------

typedef __attribute__((ext_vector_type(16))) __bf16 v16bf;
typedef __attribute__((ext_vector_type(8)))  float  v8f;

#define N_NODES    100000
#define N_EDGES    3200000
#define IN_CH      128
#define HID        128
#define OUT_CH     64
#define NUM_GRAPHS 128
#define KDIM       128   // all GEMMs in this net have K == 128

// ---------------- small utility kernels ----------------

__global__ void zero_kernel(float* __restrict__ p, int n) {
    int i = blockIdx.x * blockDim.x + threadIdx.x;
    if (i < n) p[i] = 0.0f;
}

// deg[dst] += 1 for every edge
__global__ void degree_kernel(const int* __restrict__ dst, float* __restrict__ deg) {
    int e = blockIdx.x * blockDim.x + threadIdx.x;
    if (e < N_EDGES) atomicAdd(&deg[dst[e]], 1.0f);
}

// dinv = rsqrt(deg + 1)   (self-loop)
__global__ void dinv_kernel(float* __restrict__ deg) {
    int i = blockIdx.x * blockDim.x + threadIdx.x;
    if (i < N_NODES) deg[i] = rsqrtf(deg[i] + 1.0f);
}

// h = xw * dinv^2   (self-loop term; initializes the aggregation buffer)
__global__ void selfinit_kernel(const float* __restrict__ xw,
                                const float* __restrict__ dinv,
                                float* __restrict__ h) {
    long long i = (long long)blockIdx.x * blockDim.x + threadIdx.x;
    if (i < (long long)N_NODES * HID) {
        int node = (int)(i >> 7);          // HID == 128
        float d = dinv[node];
        h[i] = xw[i] * d * d;
    }
}

// one wave32 per edge; each lane moves a float4 feature chunk
__global__ void edge_kernel(const int* __restrict__ src,
                            const int* __restrict__ dst,
                            const float* __restrict__ dinv,
                            const float* __restrict__ xw,
                            float* __restrict__ h) {
    long long t = (long long)blockIdx.x * blockDim.x + threadIdx.x;
    long long e = t >> 5;
    if (e >= N_EDGES) return;
    int lane = (int)(t & 31);
    int s = src[e];                         // wave-uniform -> scalar load
    int d = dst[e];
    float w = dinv[s] * dinv[d];
    const float4 v = *(const float4*)(xw + (long long)s * HID + lane * 4);
    float* out = h + (long long)d * HID + lane * 4;
    atomicAdd(out + 0, v.x * w);
    atomicAdd(out + 1, v.y * w);
    atomicAdd(out + 2, v.z * w);
    atomicAdd(out + 3, v.w * w);
}

// mean-pool accumulation: sums[batch[n], f] += h[n, f], cnt[batch[n]] += 1
__global__ void pool_kernel(const float* __restrict__ h,
                            const int* __restrict__ batch,
                            float* __restrict__ sums,
                            float* __restrict__ cnt) {
    long long i = (long long)blockIdx.x * blockDim.x + threadIdx.x;
    if (i < (long long)N_NODES * HID) {
        int node = (int)(i >> 7);
        int f    = (int)(i & 127);
        int b    = batch[node];
        atomicAdd(&sums[b * HID + f], h[i]);
        if (f == 0) atomicAdd(&cnt[b], 1.0f);
    }
}

__global__ void pooldiv_kernel(float* __restrict__ sums, const float* __restrict__ cnt) {
    int i = blockIdx.x * blockDim.x + threadIdx.x;
    if (i < NUM_GRAPHS * HID) {
        int b = i >> 7;
        sums[i] = sums[i] / fmaxf(cnt[b], 1.0f);
    }
}

// ---------------- WMMA GEMM ----------------
//  C[M x (NTILES*16)] = act(A[M x 128]) @ B[128 x NTILES*16] + bias
//
// Block = 256 threads = 8 waves. The whole B matrix (<=128x128) is staged in
// LDS as bf16 in *fragment order*: frag (nt,kk) = 1024B, lane-local 32B run.
// Each wave computes a 16-row x full-width strip: per k-step it builds one A
// fragment (coalesced b128 loads + cvt_pk_bf16) and feeds it to NTILES WMMAs
// whose B fragments are single 32-byte LDS reads (ds_load_b128 x2).
//
// VGPR layouts (CDNA5 ISA 7.12.2, wave32):
//   A 16x32 bf16: lane -> row m = lane%16, half = lane/16;
//                 a[e] holds k = (e<8 ? 0 : 16) + half*8 + (e%8)
//   B 32x16 bf16: lane -> col n = lane%16, half = lane/16;
//                 b[e] holds k = (e<8 ? 0 : 16) + half*8 + (e%8)
//   C 16x16 f32:  lane -> col n = lane%16; acc[r] -> row = half*8 + r
template <int NTILES, int RELU>
__global__ __launch_bounds__(256) void gemm_wmma_kernel(
    const float* __restrict__ A, const float* __restrict__ B,
    const float* __restrict__ bias, float* __restrict__ C, int M) {
    constexpr int NCOLS = NTILES * 16;
    __shared__ __align__(32) __bf16 ldsB[NTILES * (KDIM / 32) * 512];

    // ---- cooperative staging of B into LDS (bf16, fragment order) ----
    for (int e = threadIdx.x; e < NTILES * 2048; e += blockDim.x) {
        int frag = e >> 9;                 // (nt*4 + kk)
        int rem  = e & 511;
        int lane = rem >> 4;
        int elem = rem & 15;
        int nt   = frag >> 2;
        int kk   = frag & 3;
        int k    = kk * 32 + ((elem >> 3) << 4) + (lane >> 4) * 8 + (elem & 7);
        int ncol = nt * 16 + (lane & 15);
        ldsB[e] = (__bf16)B[k * NCOLS + ncol];
    }
    __syncthreads();

    int lane  = threadIdx.x & 31;
    int mtile = blockIdx.x * 8 + (threadIdx.x >> 5);
    if (mtile >= (M >> 4)) return;         // wave-uniform exit

    int half = lane >> 4;
    int lm   = lane & 15;
    int mrow = mtile * 16 + lm;            // A row owned by this lane

    v8f acc[NTILES];
#pragma unroll
    for (int nt = 0; nt < NTILES; ++nt) acc[nt] = v8f{};

#pragma unroll
    for (int kk = 0; kk < 4; ++kk) {       // K = 128 in steps of 32
        // A fragment: coalesced loads, compile-time relu, cvt to bf16
        v16bf a;
        const float* Arow = A + (long long)mrow * KDIM + kk * 32;
#pragma unroll
        for (int p = 0; p < 8; ++p) {
            int k0 = ((p >> 2) << 4) + half * 8 + ((p & 3) << 1);
            float2 v = *(const float2*)(Arow + k0);
            if (RELU) { v.x = fmaxf(v.x, 0.0f); v.y = fmaxf(v.y, 0.0f); }
            a[2 * p]     = (__bf16)v.x;
            a[2 * p + 1] = (__bf16)v.y;
        }
        // reuse A across all column tiles; B frags are 32B LDS reads
#pragma unroll
        for (int nt = 0; nt < NTILES; ++nt) {
            const v16bf b = *(const v16bf*)&ldsB[((nt * 4 + kk) << 9) + (lane << 4)];
            acc[nt] = __builtin_amdgcn_wmma_f32_16x16x32_bf16(
                false, a, false, b, (short)0, acc[nt], false, false);
        }
    }

#pragma unroll
    for (int nt = 0; nt < NTILES; ++nt) {
        int ncol = nt * 16 + lm;
        float bval = bias ? bias[ncol] : 0.0f;
#pragma unroll
        for (int r = 0; r < 8; ++r) {
            int row = mtile * 16 + half * 8 + r;
            C[(long long)row * NCOLS + ncol] = acc[nt][r] + bval;
        }
    }
}

// ---------------- driver ----------------

static inline int cdiv(long long a, int b) { return (int)((a + b - 1) / b); }

extern "C" void kernel_launch(void* const* d_in, const int* in_sizes, int n_in,
                              void* d_out, int out_size, void* d_ws, size_t ws_size,
                              hipStream_t stream) {
    const float* x    = (const float*)d_in[0];
    const int*   ei   = (const int*)d_in[1];    // [2, E]
    const int*   bat  = (const int*)d_in[2];
    const float* W0   = (const float*)d_in[3];
    const float* W1   = (const float*)d_in[4];
    const float* W2   = (const float*)d_in[5];
    const float* M0w  = (const float*)d_in[6];
    const float* M0b  = (const float*)d_in[7];
    const float* M1w  = (const float*)d_in[8];
    const float* M1b  = (const float*)d_in[9];
    const int* src = ei;
    const int* dst = ei + N_EDGES;

    // scratch layout (~103 MB)
    const size_t SZ_NF = (size_t)N_NODES * HID * sizeof(float);   // 51.2 MB
    char* ws = (char*)d_ws;
    float* dinv = (float*)ws;                                      // N floats
    float* xw   = (float*)(ws + (512ull << 10));                   // N x 128
    float* h    = (float*)(ws + (512ull << 10) + SZ_NF);           // N x 128
    float* sums = (float*)(ws + (512ull << 10) + 2 * SZ_NF);       // G x 128
    float* cnt  = sums + NUM_GRAPHS * HID;                         // G
    float* mlp1 = cnt + NUM_GRAPHS;                                // G x 128

    const int B = 256;

    // --- degrees / symmetric norm ---
    zero_kernel<<<cdiv(N_NODES, B), B, 0, stream>>>(dinv, N_NODES);
    degree_kernel<<<cdiv(N_EDGES, B), B, 0, stream>>>(dst, dinv);
    dinv_kernel<<<cdiv(N_NODES, B), B, 0, stream>>>(dinv);

    // --- 3 GCN layers: gemm -> self-loop init -> edge scatter-add ---
    const long long nfeat = (long long)N_NODES * HID;
    const int gemm_blocks = cdiv(N_NODES / 16, 8);   // 8 waves (mtiles) / block
    for (int l = 0; l < 3; ++l) {
        if (l == 0) {
            gemm_wmma_kernel<8, 0><<<gemm_blocks, 256, 0, stream>>>(
                x, W0, nullptr, xw, N_NODES);
        } else {
            gemm_wmma_kernel<8, 1><<<gemm_blocks, 256, 0, stream>>>(
                h, (l == 1) ? W1 : W2, nullptr, xw, N_NODES);
        }
        selfinit_kernel<<<cdiv(nfeat, B), B, 0, stream>>>(xw, dinv, h);
        edge_kernel<<<cdiv((long long)N_EDGES * 32, B), B, 0, stream>>>(src, dst, dinv, xw, h);
    }

    // --- global mean pool ---
    zero_kernel<<<cdiv(NUM_GRAPHS * HID + NUM_GRAPHS, B), B, 0, stream>>>(
        sums, NUM_GRAPHS * HID + NUM_GRAPHS);
    pool_kernel<<<cdiv(nfeat, B), B, 0, stream>>>(h, bat, sums, cnt);
    pooldiv_kernel<<<cdiv(NUM_GRAPHS * HID, B), B, 0, stream>>>(sums, cnt);

    // --- MLP head (same WMMA gemm; M = 128 -> one block of 8 waves) ---
    gemm_wmma_kernel<8, 0><<<cdiv(NUM_GRAPHS / 16, 8), 256, 0, stream>>>(
        sums, M0w, M0b, mlp1, NUM_GRAPHS);
    gemm_wmma_kernel<4, 1><<<cdiv(NUM_GRAPHS / 16, 8), 256, 0, stream>>>(
        mlp1, M1w, M1b, (float*)d_out, NUM_GRAPHS);
}
